// ActorModel_61074434949879
// MI455X (gfx1250) — matrix operand order repaired and checked by reading, not compiled, not run on hardware
//
#include <hip/hip_runtime.h>

#define HIDN    548
#define LSTMIN  224
#define G4      2192
#define GPAD    2208     // gate rows padded to 16*138 (zero rows) -> no row guards
#define WHS     576      // Whh/Wout K padded to multiple of 32
#define SHW     584      // LDS row stride (halves) for h0/h_out (bank-conflict-free)
#define SCW     232      // LDS row stride (halves) for combined staging
#define PHASES  8
#define NW      8        // waves per workgroup (lockstep weight streaming through WGP$)

typedef __attribute__((ext_vector_type(16))) _Float16     v16h;
typedef __attribute__((ext_vector_type(8)))  float        v8f;
typedef __attribute__((ext_vector_type(4)))  unsigned int v4u;
typedef __attribute__((ext_vector_type(8)))  unsigned int v8u;

__device__ __forceinline__ v8f wmma_f16(v16h a, v16h b, v8f c) {
  // D = A(16x32 f16) x B(32x16 f16) + C(16x16 f32)
  return __builtin_amdgcn_wmma_f32_16x16x32_f16(false, a, false, b, (short)0, c, false, false);
}

__device__ __forceinline__ v16h pack_frag(const _Float16* p0, const _Float16* p1) {
  v4u a = *reinterpret_cast<const v4u*>(p0);
  v4u b = *reinterpret_cast<const v4u*>(p1);
  v8u u = __builtin_shufflevector(a, b, 0, 1, 2, 3, 4, 5, 6, 7);
  return __builtin_bit_cast(v16h, u);
}

// A-matrix 16x32 f16 fragment from LDS (row-major, stride in halves).
// Lane layout (ISA 7.12.2): lanes 0-15 row m=lane, K {0..7,16..23}; lanes 16-31 K {8..15,24..31}.
__device__ __forceinline__ v16h load_a_lds(const _Float16* base, int strideH, int k0) {
  const int lane = threadIdx.x & 31;
  const int m = lane & 15, hi = lane >> 4;
  const _Float16* p = base + m * strideH + k0 + 8 * hi;   // 16B aligned
  return pack_frag(p, p + 16);
}

// B-matrix 32x16 f16 fragment from global weights stored as W[row][k] row-major:
// lane n = column = weight row; lanes 0-15 K {0..7,16..23}, lanes 16-31 K {8..15,24..31} (matches A).
__device__ __forceinline__ v16h load_b_glb(const _Float16* __restrict__ W, int strideH,
                                           int rowBase, int k0) {
  const int lane = threadIdx.x & 31;
  const int n = lane & 15, hi = lane >> 4;
  const _Float16* p = W + (size_t)(rowBase + n) * strideH + k0 + 8 * hi;
  return pack_frag(p, p + 16);
}

struct B4 { v16h i, f, g, o; };
__device__ __forceinline__ B4 load_b4(const _Float16* __restrict__ W, int strideH,
                                      int ri, int rf, int rg, int ro, int k0) {
  B4 r;
  r.i = load_b_glb(W, strideH, ri, k0);
  r.f = load_b_glb(W, strideH, rf, k0);
  r.g = load_b_glb(W, strideH, rg, k0);
  r.o = load_b_glb(W, strideH, ro, k0);
  return r;
}

__device__ __forceinline__ float sigm(float x)   { return 1.f / (1.f + __expf(-x)); }
__device__ __forceinline__ float tanh_f(float x) { return 1.f - 2.f / (1.f + __expf(2.f * x)); }

// ---------------- weight preparation: f32 -> padded f16 in d_ws ----------------
__global__ void prep_weights(const float* __restrict__ Wih, const float* __restrict__ Whh,
                             const float* __restrict__ Wout,
                             const float* __restrict__ bih, const float* __restrict__ bhh,
                             _Float16* __restrict__ Wihh, _Float16* __restrict__ Whhh,
                             _Float16* __restrict__ Wouth, float* __restrict__ bsum) {
  const int stride = gridDim.x * blockDim.x;
  const int tid = blockIdx.x * blockDim.x + threadIdx.x;
  for (int idx = tid; idx < GPAD * LSTMIN; idx += stride) {
    int r = idx / LSTMIN, c = idx - r * LSTMIN;
    Wihh[idx] = (r < G4) ? (_Float16)Wih[(size_t)r * LSTMIN + c] : (_Float16)0.f;
  }
  for (int idx = tid; idx < GPAD * WHS; idx += stride) {
    int r = idx / WHS, c = idx - r * WHS;
    Whhh[idx] = (r < G4 && c < HIDN) ? (_Float16)Whh[(size_t)r * HIDN + c] : (_Float16)0.f;
  }
  for (int idx = tid; idx < 16 * WHS; idx += stride) {
    int r = idx / WHS, c = idx - r * WHS;
    Wouth[idx] = (r < PHASES && c < HIDN) ? (_Float16)Wout[(size_t)r * HIDN + c] : (_Float16)0.f;
  }
  for (int idx = tid; idx < GPAD; idx += stride)
    bsum[idx] = (idx < G4) ? (bih[idx] + bhh[idx]) : 0.f;
}

// ---------------- fully fused branches + LSTM cell + head + softmax ----------------
// One wave owns 16 batch rows; NW waves per workgroup march through the same
// weight stream in lockstep (barrier per hidden tile) so WGP$ captures the reuse.
__global__ __launch_bounds__(256) void actor_fused(
    const float* __restrict__ wave, const float* __restrict__ waitv,
    const float* __restrict__ neigh,
    const float* __restrict__ W1, const float* __restrict__ b1,
    const float* __restrict__ W2, const float* __restrict__ b2,
    const float* __restrict__ W3, const float* __restrict__ b3,
    const float* __restrict__ bout,
    const float* __restrict__ h0, const float* __restrict__ c0,
    const _Float16* __restrict__ Wihh, const _Float16* __restrict__ Whhh,
    const _Float16* __restrict__ Wouth, const float* __restrict__ bsum,
    float* __restrict__ out) {
  extern __shared__ _Float16 smem[];                 // NW * 2 * 16*SHW halves
  const int lane  = threadIdx.x & 31;
  const int wid   = threadIdx.x >> 5;
  const int row0  = blockIdx.x * (16 * NW) + wid * 16;
  const int n     = lane & 15;        // C/D column within tile
  const int mBase = (lane >> 4) * 8;  // C/D row base for this lane half

  _Float16* mH = smem + (size_t)wid * (2 * 16 * SHW);  // f16 h0 (K padded to 576)
  _Float16* mO = mH + 16 * SHW;                        // combined staging, then f16 h_out

  // ---- stage h0 -> f16 LDS, zero-padded to 576 ----
  for (int m = 0; m < 16; ++m) {
    const float* src = h0 + (size_t)(row0 + m) * HIDN;
    for (int c = lane; c < SHW; c += 32)
      mH[m * SHW + c] = (c < HIDN) ? (_Float16)src[c] : (_Float16)0.f;
  }

  // ---- branch FCs -> combined[16][224] f16 into mO (stride SCW) ----
  for (int col = lane; col < 128; col += 32) {        // x1 = wave @ W1^T + b1
    float w[12];
#pragma unroll
    for (int k = 0; k < 12; ++k) w[k] = W1[col * 12 + k];
    const float bias = b1[col];
    for (int m = 0; m < 16; ++m) {
      const float* in = wave + (size_t)(row0 + m) * 12;
      float s = bias;
#pragma unroll
      for (int k = 0; k < 12; ++k) s += in[k] * w[k];
      mO[m * SCW + col] = (_Float16)s;
    }
  }
  {                                                   // x2 = wait @ W2^T + b2 (32 cols)
    const int c2 = lane;
    float w[12];
#pragma unroll
    for (int k = 0; k < 12; ++k) w[k] = W2[c2 * 12 + k];
    const float bias = b2[c2];
    for (int m = 0; m < 16; ++m) {
      const float* in = waitv + (size_t)(row0 + m) * 12;
      float s = bias;
#pragma unroll
      for (int k = 0; k < 12; ++k) s += in[k] * w[k];
      mO[m * SCW + 128 + c2] = (_Float16)s;
    }
  }
  for (int cc = lane; cc < 64; cc += 32) {            // x3 = neigh @ W3^T + b3
    float w[48];
#pragma unroll
    for (int k = 0; k < 48; ++k) w[k] = W3[cc * 48 + k];
    const float bias = b3[cc];
    for (int m = 0; m < 16; ++m) {
      const float* in = neigh + (size_t)(row0 + m) * 48;
      float s = bias;
#pragma unroll
      for (int k = 0; k < 48; ++k) s += in[k] * w[k];
      mO[m * SCW + 160 + cc] = (_Float16)s;
    }
  }
  __syncthreads();

  // ---- cache the 7 combined A-fragments in VGPRs (reused by all 35 hidden tiles) ----
  v16h Awih[7];
#pragma unroll
  for (int ks = 0; ks < 7; ++ks) Awih[ks] = load_a_lds(mO, SCW, ks * 32);

  // ---- gate GEMMs + LSTM nonlinearity per 16-wide hidden tile ----
#pragma unroll 1
  for (int t = 0; t < 35; ++t) {
    __syncthreads();   // keep all 8 waves on the same weight tile (WGP$ reuse)
    const int ri = t * 16, rf = 548 + t * 16, rg = 1096 + t * 16, ro = 1644 + t * 16;
    const int hu = t * 16 + n;
    const bool vh = hu < HIDN;
    // hoist epilogue operands so their latency hides under the WMMAs
    const float bi = bsum[ri + n], bf = bsum[rf + n];
    const float bg = bsum[rg + n], bo = bsum[ro + n];
    float c0v[8];
#pragma unroll
    for (int j = 0; j < 8; ++j)
      c0v[j] = vh ? c0[(size_t)(row0 + mBase + j) * HIDN + hu] : 0.f;
    if (t + 1 < 35) {  // nudge next tile's weight rows toward the caches
      __builtin_prefetch(Wihh + (size_t)(ri + 16) * LSTMIN, 0, 3);
      __builtin_prefetch(Wihh + (size_t)(rf + 16) * LSTMIN, 0, 3);
      __builtin_prefetch(Wihh + (size_t)(rg + 16) * LSTMIN, 0, 3);
      __builtin_prefetch(Wihh + (size_t)(ro + 16) * LSTMIN, 0, 3);
    }

    v8f ai = {0.f, 0.f, 0.f, 0.f, 0.f, 0.f, 0.f, 0.f};
    v8f af = ai, ag = ai, ao = ai;

    // -- software-pipelined B fragments: load ks+1 before consuming ks --
    B4 cur = load_b4(Wihh, LSTMIN, ri, rf, rg, ro, 0);
#pragma unroll
    for (int ks = 0; ks < 7; ++ks) {                 // combined @ Wih^T
      B4 nxt = (ks < 6) ? load_b4(Wihh, LSTMIN, ri, rf, rg, ro, (ks + 1) * 32)
                        : load_b4(Whhh, WHS, ri, rf, rg, ro, 0);  // handoff prefetch
      const v16h a = Awih[ks];
      ai = wmma_f16(a, cur.i, ai);
      af = wmma_f16(a, cur.f, af);
      ag = wmma_f16(a, cur.g, ag);
      ao = wmma_f16(a, cur.o, ao);
      cur = nxt;
    }
#pragma unroll 2
    for (int ks = 0; ks < 17; ++ks) {                // h0 @ Whh^T (K padded to 576)
      B4 nxt = load_b4(Whhh, WHS, ri, rf, rg, ro, (ks + 1) * 32);
      const v16h a = load_a_lds(mH, SHW, ks * 32);
      ai = wmma_f16(a, cur.i, ai);
      af = wmma_f16(a, cur.f, af);
      ag = wmma_f16(a, cur.g, ag);
      ao = wmma_f16(a, cur.o, ao);
      cur = nxt;
    }
    {                                                // peeled last Whh step
      const v16h a = load_a_lds(mH, SHW, 17 * 32);
      ai = wmma_f16(a, cur.i, ai);
      af = wmma_f16(a, cur.f, af);
      ag = wmma_f16(a, cur.g, ag);
      ao = wmma_f16(a, cur.o, ao);
    }

#pragma unroll
    for (int j = 0; j < 8; ++j) {
      const int m = mBase + j;
      const float iv = sigm(ai[j] + bi);
      const float fv = sigm(af[j] + bf);
      const float gv = tanh_f(ag[j] + bg);
      const float ov = sigm(ao[j] + bo);
      const float cv = fv * c0v[j] + iv * gv;
      const float hv = ov * tanh_f(cv);
      mO[m * SHW + hu] = vh ? (_Float16)hv : (_Float16)0.f;
    }
  }

  // zero h_out pad columns [560,576) so the head K-loop reads zeros there
  for (int m = 0; m < 16; ++m) {
    const int c = 560 + lane;
    if (c < WHS) mO[m * SHW + c] = (_Float16)0.f;
  }
  __syncthreads();

  // ---- head: logits = h @ Wout^T (N padded 8->16, K padded 548->576) ----
  v8f al = {0.f, 0.f, 0.f, 0.f, 0.f, 0.f, 0.f, 0.f};
  v16h bcur = load_b_glb(Wouth, WHS, 0, 0);
#pragma unroll 2
  for (int ks = 0; ks < 17; ++ks) {
    v16h bnxt = load_b_glb(Wouth, WHS, 0, (ks + 1) * 32);
    al = wmma_f16(load_a_lds(mO, SHW, ks * 32), bcur, al);
    bcur = bnxt;
  }
  al = wmma_f16(load_a_lds(mO, SHW, 17 * 32), bcur, al);

  const float bph = (n < PHASES) ? bout[n] : 0.f;
#pragma unroll
  for (int j = 0; j < 8; ++j) {
    const float v = al[j] + bph;
    float mx = (n < PHASES) ? v : -3.0e38f;          // softmax over the 8 phase lanes
    mx = fmaxf(mx, __shfl_xor(mx, 1, 32));
    mx = fmaxf(mx, __shfl_xor(mx, 2, 32));
    mx = fmaxf(mx, __shfl_xor(mx, 4, 32));
    float e = (n < PHASES) ? __expf(v - mx) : 0.f;
    float s = e;
    s += __shfl_xor(s, 1, 32);
    s += __shfl_xor(s, 2, 32);
    s += __shfl_xor(s, 4, 32);
    if (n < PHASES)
      out[(size_t)(row0 + mBase + j) * PHASES + n] = e / s;
  }
}

extern "C" void kernel_launch(void* const* d_in, const int* in_sizes, int n_in,
                              void* d_out, int out_size, void* d_ws, size_t ws_size,
                              hipStream_t stream) {
  (void)n_in; (void)out_size; (void)ws_size;
  const float* wave  = (const float*)d_in[0];
  const float* waitv = (const float*)d_in[1];
  const float* neigh = (const float*)d_in[2];
  const float* W1    = (const float*)d_in[3];
  const float* b1    = (const float*)d_in[4];
  const float* W2    = (const float*)d_in[5];
  const float* b2    = (const float*)d_in[6];
  const float* W3    = (const float*)d_in[7];
  const float* b3    = (const float*)d_in[8];
  const float* Wih   = (const float*)d_in[9];
  const float* Whh   = (const float*)d_in[10];
  const float* bih   = (const float*)d_in[11];
  const float* bhh   = (const float*)d_in[12];
  const float* Wout  = (const float*)d_in[13];
  const float* bout  = (const float*)d_in[14];
  const float* h0    = (const float*)d_in[15];
  const float* c0    = (const float*)d_in[16];
  float* out = (float*)d_out;

  _Float16* Wihh  = (_Float16*)d_ws;                       // [2208][224]
  _Float16* Whhh  = Wihh + (size_t)GPAD * LSTMIN;          // [2208][576]
  _Float16* Wouth = Whhh + (size_t)GPAD * WHS;             // [16][576]
  float*    bsum  = (float*)(Wouth + 16 * WHS);            // [2208]

  const int batch = in_sizes[0] / 12;                      // 65536
  const size_t shmem = (size_t)NW * 2 * 16 * SHW * sizeof(_Float16);  // ~292 KB of 320 KB

  prep_weights<<<2048, 256, 0, stream>>>(Wih, Whh, Wout, bih, bhh, Wihh, Whhh, Wouth, bsum);
  actor_fused<<<batch / (16 * NW), 256, shmem, stream>>>(wave, waitv, neigh, W1, b1, W2, b2,
                                                         W3, b3, bout, h0, c0,
                                                         Wihh, Whhh, Wouth, bsum, out);
}
